// TextEncoder_24859270709534
// MI455X (gfx1250) — matrix operand order repaired and checked by reading, not compiled, not run on hardware
//
#include <hip/hip_runtime.h>
#include <hip/hip_bf16.h>
#include <stdint.h>

typedef __attribute__((ext_vector_type(16))) _Float16 v16h;
typedef __attribute__((ext_vector_type(8)))  _Float16 v8h;
typedef __attribute__((ext_vector_type(8)))  float    v8f;

#define Bsz 256
#define Tsz 512
#define Dd  64
#define Hh  128

// ---------------------------------------------------------------------------
// WMMA helper:  D = A(16x32 f16) * B(32x16 f16) + C(16x16 f32)
// ---------------------------------------------------------------------------
__device__ __forceinline__ v8f wmma_f16(v16h a, v16h b, v8f c) {
  return __builtin_amdgcn_wmma_f32_16x16x32_f16(
      /*neg_a=*/false, a, /*neg_b=*/false, b,
      /*c_mod=*/(short)0, c, /*reuse_a=*/false, /*reuse_b=*/false);
}

// Fast 1/x via v_rcp_f32 (inf -> 0 saturation makes the gate math clamp-free).
__device__ __forceinline__ float fast_rcp(float x) {
  return __builtin_amdgcn_rcpf(x);
}

// A-fragment (16x32, f16) from an LDS tile laid out [16 rows x ldk halves].
// ISA 7.12.2: lane L (m=L&15, g=L>>4), half i -> k = (i&7) + ((i>>3)<<4) + g*8
// => two contiguous 16B chunks per lane at half-offsets (g*8) and (g*8+16).
__device__ __forceinline__ v16h load_A(const _Float16* base, int ldk, int kbase, int lane) {
  int m = lane & 15, g = lane >> 4;
  const _Float16* p = base + m * ldk + kbase + g * 8;
  v8h lo = *(const v8h*)(p);
  v8h hi = *(const v8h*)(p + 16);
  v16h r;
#pragma unroll
  for (int i = 0; i < 8; ++i) { r[i] = lo[i]; r[i + 8] = hi[i]; }
  return r;
}

// Async copy global -> LDS, 8 bytes per active lane, tracked by ASYNCcnt.
__device__ __forceinline__ void async_load_b64(void* lds_dst, const void* gsrc) {
  unsigned lds_off = (unsigned)(uintptr_t)lds_dst;  // low 32 bits = LDS byte offset
  asm volatile("global_load_async_to_lds_b64 %0, %1, off"
               :: "v"(lds_off), "v"(gsrc) : "memory");
}
__device__ __forceinline__ void wait_async0() {
  asm volatile("s_wait_asynccnt 0x0" ::: "memory");
}

// ---------------------------------------------------------------------------
// Phase 1: embedding gather  [B,T] tokens -> e_ws[T, B, D] in f16
// ---------------------------------------------------------------------------
__global__ __launch_bounds__(256)
void gather_embed_kernel(const int* __restrict__ x, const float* __restrict__ emb,
                         _Float16* __restrict__ e_ws) {
  int idx = blockIdx.x * 256 + threadIdx.x;   // over B*T*D
  int r = idx >> 6;                            // bt row (b*T + t)
  int d = idx & 63;
  int b = r >> 9;                              // T = 512
  int t = r & 511;
  int tok = x[r];
  e_ws[((size_t)t * Bsz + b) * Dd + d] = (_Float16)emb[(size_t)tok * Dd + d];
}

// ---------------------------------------------------------------------------
// Phase 2: persistent per-batch-strip GRU scan + fused input projection + LN.
// grid = 16 workgroups (one 16-row batch strip each), 256 threads = 8 waves.
// Wave j owns gate columns [16j, 16j+16) of H; weight B-fragments stay in
// VGPRs for the whole 512-step scan; e-slabs stream in via async-to-LDS; the
// lane's own slice of h stays in f32 registers (LDS copy is write-only f16,
// consumed only by other waves' A-fragment loads).
// ---------------------------------------------------------------------------
__global__ __launch_bounds__(256, 1)
void gru_scan_kernel(const _Float16* __restrict__ e_ws,
                     const float* __restrict__ w_ih, const float* __restrict__ w_hh,
                     const float* __restrict__ b_ih, const float* __restrict__ b_hh,
                     const float* __restrict__ gamma, const float* __restrict__ beta,
                     float* __restrict__ out) {
  __shared__ __align__(16) _Float16 hbuf[2][16 * Hh];   // ping-pong hidden strip
  __shared__ __align__(16) _Float16 esl[2][16 * Dd];    // ping-pong embedding slab

  const int tid   = threadIdx.x;
  const int wave  = tid >> 5;          // 0..7 -> H-column tile j
  const int lane  = tid & 31;
  const int n     = lane & 15;
  const int gl    = lane >> 4;
  const int mbase = blockIdx.x * 16;   // batch strip base
  const int hcol  = wave * 16 + n;     // this lane's output column within H

  // --- Preload weight B-fragments into registers (f32 -> f16), once. ---
  // B 32x16 layout: lane L (n=L&15, g=L>>4), half i -> k = kbase + 8*g + (i&7) + 16*(i>>3).
  v16h Bh[3][4];   // w_hh: 3 gates x 4 k-steps (K=128)
  v16h Be[3][2];   // w_ih: 3 gates x 2 k-steps (K=64)
#pragma unroll
  for (int gate = 0; gate < 3; ++gate) {
    const float* wh = w_hh + (size_t)(gate * Hh + hcol) * Hh;
    const float* wi = w_ih + (size_t)(gate * Hh + hcol) * Dd;
#pragma unroll
    for (int ks = 0; ks < 4; ++ks)
#pragma unroll
      for (int i = 0; i < 16; ++i)
        Bh[gate][ks][i] = (_Float16)wh[ks * 32 + gl * 8 + (i & 7) + ((i >> 3) << 4)];
#pragma unroll
    for (int ks = 0; ks < 2; ++ks)
#pragma unroll
      for (int i = 0; i < 16; ++i)
        Be[gate][ks][i] = (_Float16)wi[ks * 32 + gl * 8 + (i & 7) + ((i >> 3) << 4)];
  }
  // Per-lane gate biases (column-constant across the C fragment rows).
  const float brz_r = b_ih[hcol]           + b_hh[hcol];
  const float brz_z = b_ih[Hh + hcol]      + b_hh[Hh + hcol];
  const float bi_n  = b_ih[2 * Hh + hcol];
  const float bh_n  = b_hh[2 * Hh + hcol];

  // This lane's slice of the hidden state, kept in f32 registers for the
  // whole scan: rows m = r + 8*gl of the strip, column hcol.
  float hprev[8];
#pragma unroll
  for (int r = 0; r < 8; ++r) hprev[r] = 0.f;

  // h0 = 0 in LDS
  for (int i = tid; i < 16 * Hh; i += 256) hbuf[0][i] = (_Float16)0.f;

  // Kick off async fill of slab 0 (8 B per lane, 2 KB total).
  {
    const char* esrc = (const char*)(e_ws + (size_t)mbase * Dd);
    async_load_b64((char*)&esl[0][0] + tid * 8, esrc + tid * 8);
  }
  __syncthreads();

  for (int t = 0; t < Tsz; ++t) {
    const _Float16* hcur = hbuf[t & 1];
    _Float16*       hnxt = hbuf[(t + 1) & 1];
    const _Float16* ecur = esl[t & 1];

    wait_async0();        // my wave's slab portion has landed in LDS
    __syncthreads();      // everyone's portion has landed

    // Async prefetch of next step's slab into the other buffer.
    if (t + 1 < Tsz) {
      const char* esrc = (const char*)(e_ws + ((size_t)(t + 1) * Bsz + mbase) * Dd);
      async_load_b64((char*)&esl[(t + 1) & 1][0] + tid * 8, esrc + tid * 8);
    }

    // Accumulators: r/z combine both sides (+both biases); n-gate kept split.
    v8f c_r, c_z, c_in, c_hn;
#pragma unroll
    for (int r = 0; r < 8; ++r) { c_r[r] = brz_r; c_z[r] = brz_z; c_in[r] = bi_n; c_hn[r] = bh_n; }

    // h-side GEMM: K = 128 -> 4 k-steps, A shared across the three gates.
#pragma unroll
    for (int ks = 0; ks < 4; ++ks) {
      v16h a = load_A(hcur, Hh, ks * 32, lane);
      c_r  = wmma_f16(a, Bh[0][ks], c_r);
      c_z  = wmma_f16(a, Bh[1][ks], c_z);
      c_hn = wmma_f16(a, Bh[2][ks], c_hn);
    }
    // input-side GEMM (fused, no materialized gi): K = 64 -> 2 k-steps.
#pragma unroll
    for (int ks = 0; ks < 2; ++ks) {
      v16h a = load_A(ecur, Dd, ks * 32, lane);
      c_r  = wmma_f16(a, Be[0][ks], c_r);
      c_z  = wmma_f16(a, Be[1][ks], c_z);
      c_in = wmma_f16(a, Be[2][ks], c_in);
    }

    // Gate nonlinearity + state update. C layout: row m = r + 8*gl, col = hcol.
    // sigmoid(x) = rcp(1+exp(-x)); tanh(x) = 1 - 2*rcp(exp(2x)+1)  (v_rcp_f32).
    // h_old comes from this lane's own registers (it wrote the same (m,hcol)
    // slots last step), so no LDS read-back on the critical path.
#pragma unroll
    for (int r = 0; r < 8; ++r) {
      int m = r + 8 * gl;
      float rr = fast_rcp(1.f + __expf(-c_r[r]));
      float zz = fast_rcp(1.f + __expf(-c_z[r]));
      float pre = c_in[r] + rr * c_hn[r];
      float nn = 1.f - 2.f * fast_rcp(__expf(2.f * pre) + 1.f);
      float hnew = (1.f - zz) * nn + zz * hprev[r];
      hprev[r] = hnew;
      hnxt[m * Hh + hcol] = (_Float16)hnew;
    }
    __syncthreads();
  }

  // --- Final LayerNorm on this strip's h_T (lives in hbuf[Tsz & 1]). ---
  const _Float16* hf = hbuf[Tsz & 1];
  for (int row = wave; row < 16; row += 8) {
    float v[4];
#pragma unroll
    for (int i = 0; i < 4; ++i) v[i] = (float)hf[row * Hh + lane * 4 + i];
    float s = v[0] + v[1] + v[2] + v[3];
#pragma unroll
    for (int off = 16; off >= 1; off >>= 1) s += __shfl_xor(s, off, 32);
    float mu = s * (1.f / 128.f);
    float d2 = 0.f;
#pragma unroll
    for (int i = 0; i < 4; ++i) { float d = v[i] - mu; d2 += d * d; }
#pragma unroll
    for (int off = 16; off >= 1; off >>= 1) d2 += __shfl_xor(d2, off, 32);
    float inv = rsqrtf(d2 * (1.f / 128.f) + 1e-5f);
#pragma unroll
    for (int i = 0; i < 4; ++i) {
      int c = lane * 4 + i;
      out[(size_t)(mbase + row) * Hh + c] = (v[i] - mu) * inv * gamma[c] + beta[c];
    }
  }
}

// ---------------------------------------------------------------------------
extern "C" void kernel_launch(void* const* d_in, const int* in_sizes, int n_in,
                              void* d_out, int out_size, void* d_ws, size_t ws_size,
                              hipStream_t stream) {
  const int*   x     = (const int*)d_in[0];
  const float* emb   = (const float*)d_in[1];
  const float* w_ih  = (const float*)d_in[2];
  const float* w_hh  = (const float*)d_in[3];
  const float* b_ih  = (const float*)d_in[4];
  const float* b_hh  = (const float*)d_in[5];
  const float* gamma = (const float*)d_in[6];
  const float* beta  = (const float*)d_in[7];
  float*       out   = (float*)d_out;
  _Float16*    e_ws  = (_Float16*)d_ws;      // [T, B, D] f16 staging (16 MB)

  // Phase 1: gather + fp32->fp16 convert, time-major for contiguous per-step slabs.
  gather_embed_kernel<<<(Bsz * Tsz * Dd) / 256, 256, 0, stream>>>(x, emb, e_ws);

  // Phase 2: 16 independent batch strips, each scans all 512 steps locally.
  gru_scan_kernel<<<16, 256, 0, stream>>>(e_ws, w_ih, w_hh, b_ih, b_hh, gamma, beta, out);
}
